// PostModel_73650099192257
// MI455X (gfx1250) — matrix operand order repaired and checked by reading, not compiled, not run on hardware
//
#include <hip/hip_runtime.h>
#include <hip/hip_bf16.h>

typedef __attribute__((ext_vector_type(16))) _Float16 v16h;
typedef __attribute__((ext_vector_type(8)))  float    v8f;

#define BATCH   16
#define NCLS    20
#define NA_TOT  8400    // per-batch decoded anchors: 6400 + 1600 + 400
#define NCAND   2400    // after prefilter: 1000 + 1000 + 400
#define TOPK    100
#define IOU_T   0.5f
#define SCORE_T 0.3f
#define IMGSZ   640.0f

// ---------------------------------------------------------------------------
// CDNA5 async global->LDS staging (ASYNCcnt-tracked; ISA §15.18 / §10.x).
// LDS byte offset = low 32 bits of the generic (LDS-aperture) address.
// ---------------------------------------------------------------------------
__device__ __forceinline__ void async_g2lds_b128(void* lds_ptr, const void* gptr) {
  unsigned l = (unsigned)(size_t)lds_ptr;
  asm volatile("global_load_async_to_lds_b128 %0, %1, off"
               :: "v"(l), "v"(gptr) : "memory");
}
__device__ __forceinline__ void async_g2lds_b32(void* lds_ptr, const void* gptr) {
  unsigned l = (unsigned)(size_t)lds_ptr;
  asm volatile("global_load_async_to_lds_b32 %0, %1, off"
               :: "v"(l), "v"(gptr) : "memory");
}
__device__ __forceinline__ void wait_async0() {
  asm volatile("s_wait_asynccnt 0x0" ::: "memory");
}

// ---------------------------------------------------------------------------
// Kernel 1: fused per-level decode.
//   - sigmoid class scores + per-anchor max score (for prefilter key)
//   - DFL distance decode via WMMA: one v_wmma_f32_16x16x32_f16 computes
//     numerator (sum p*k) and denominator (sum p) for 4 sides x 16 anchors.
//   - distance2bbox with clipping
// One wave handles 16 anchors. 8 waves / block.
// ---------------------------------------------------------------------------
__global__ __launch_bounds__(256) void decode_level_kernel(
    const float* __restrict__ cls, const float* __restrict__ bbox,
    float* __restrict__ dboxes, float* __restrict__ dscores,
    float* __restrict__ dmax,
    int h, int w, float stride, int lvl_off, int groups)
{
  __shared__ __align__(16) float dred[8][16][8];   // per-wave D staging
  const int tid  = threadIdx.x;
  const int wid  = tid >> 5;
  const int lane = tid & 31;
  const int half = lane >> 4;      // which 16-lane half of the wave
  const int m    = lane & 15;
  const int wave = blockIdx.x * 8 + wid;
  const int b    = wave / groups;
  const int g    = wave % groups;
  const int n    = h * w;
  const int a    = g * 16 + m;     // anchor handled by this lane

  // Prefetch the class-score line we will need after the WMMA phase.
  __builtin_prefetch(cls + ((size_t)b * n + a) * NCLS, 0, 1);

  // ---- B fragment (constant DFL weights), K=32 x N=16, 16-bit layout:
  //   lanes 0-15 hold K=0-7,16-23; lanes 16-31 hold K=8-15,24-31; column N = lane&15.
  //   col j in 0..3 : weight = bin index t for side j   (numerator)
  //   col j in 4..7 : weight = 1 for side j-4            (denominator)
  v16h bfrag;
#pragma unroll
  for (int e = 0; e < 16; ++e) {
    const int K = 8 * half + e + (e >= 8 ? 8 : 0);
    const int j = K >> 3, t = K & 7;
    float v = 0.0f;
    if (m == j)          v = (float)t;
    else if (m == j + 4) v = 1.0f;
    bfrag[e] = (_Float16)v;
  }

  // ---- A fragment: exp(x - max) per 8-bin side. 16-bit A layout (16x32):
  //   lanes 0-15:  elems 0-7 = K 0-7,   elems 8-15 = K 16-23
  //   lanes 16-31: elems 0-7 = K 8-15,  elems 8-15 = K 24-31
  // So each lane owns two complete sides (half, half+2) of its anchor.
  const float* bb = bbox + ((size_t)b * n + a) * 32;
  const int s0 = half, s1 = half + 2;
  float xa[8], xb[8];
  {
    float4 p0 = *(const float4*)(bb + 8 * s0);
    float4 p1 = *(const float4*)(bb + 8 * s0 + 4);
    xa[0]=p0.x; xa[1]=p0.y; xa[2]=p0.z; xa[3]=p0.w;
    xa[4]=p1.x; xa[5]=p1.y; xa[6]=p1.z; xa[7]=p1.w;
    float4 q0 = *(const float4*)(bb + 8 * s1);
    float4 q1 = *(const float4*)(bb + 8 * s1 + 4);
    xb[0]=q0.x; xb[1]=q0.y; xb[2]=q0.z; xb[3]=q0.w;
    xb[4]=q1.x; xb[5]=q1.y; xb[6]=q1.z; xb[7]=q1.w;
  }
  float mxa = xa[0], mxb = xb[0];
#pragma unroll
  for (int e = 1; e < 8; ++e) { mxa = fmaxf(mxa, xa[e]); mxb = fmaxf(mxb, xb[e]); }
  v16h afrag;
#pragma unroll
  for (int e = 0; e < 8; ++e) {
    afrag[e]     = (_Float16)__expf(xa[e] - mxa);
    afrag[e + 8] = (_Float16)__expf(xb[e] - mxb);
  }

  v8f acc = {};
  acc = __builtin_amdgcn_wmma_f32_16x16x32_f16(
      /*neg_a=*/false, afrag, /*neg_b=*/false, bfrag,
      /*c_mod=*/(short)0, acc, /*reuse_a=*/false, /*reuse_b=*/false);

  // D layout: lanes 0-15 N=lane, VGPR r -> M=r; lanes 16-31 N=lane-16, M=r+8.
  if (m < 8) {
#pragma unroll
    for (int r = 0; r < 8; ++r) dred[wid][r + 8 * half][m] = acc[r];
  }
  __syncthreads();

  if (half == 0) {
    // lanes 0-15: boxes for their anchor
    const float d0 = dred[wid][m][0] / dred[wid][m][4] * stride;
    const float d1 = dred[wid][m][1] / dred[wid][m][5] * stride;
    const float d2 = dred[wid][m][2] / dred[wid][m][6] * stride;
    const float d3 = dred[wid][m][3] / dred[wid][m][7] * stride;
    const int   row = a / w, col = a - row * w;
    const float py = (row + 0.5f) * stride, px = (col + 0.5f) * stride;
    float4 o;
    o.x = fminf(fmaxf(py - d0, 0.0f), IMGSZ);   // y1
    o.y = fminf(fmaxf(px - d1, 0.0f), IMGSZ);   // x1
    o.z = fminf(fmaxf(py + d2, 0.0f), IMGSZ);   // y2
    o.w = fminf(fmaxf(px + d3, 0.0f), IMGSZ);   // x2
    *(float4*)(dboxes + ((size_t)b * NA_TOT + lvl_off + a) * 4) = o;
  } else {
    // lanes 16-31: sigmoid scores + max score for the same anchor
    const float* cp = cls + ((size_t)b * n + a) * NCLS;
    float* sp = dscores + ((size_t)b * NA_TOT + lvl_off + a) * NCLS;
    float mx = -__builtin_inff();
#pragma unroll
    for (int c4 = 0; c4 < 5; ++c4) {
      float4 q = ((const float4*)cp)[c4];
      float4 r;
      r.x = 1.0f / (1.0f + __expf(-q.x));
      r.y = 1.0f / (1.0f + __expf(-q.y));
      r.z = 1.0f / (1.0f + __expf(-q.z));
      r.w = 1.0f / (1.0f + __expf(-q.w));
      mx = fmaxf(mx, fmaxf(fmaxf(r.x, r.y), fmaxf(r.z, r.w)));
      ((float4*)sp)[c4] = r;
    }
    dmax[(size_t)b * NA_TOT + lvl_off + a] = mx;
  }
}

// ---------------------------------------------------------------------------
// In-LDS bitonic sort, descending by 64-bit key (score-bits | inverted index
// => descending score, ascending index on ties: matches jax top_k/argmax).
// ---------------------------------------------------------------------------
__device__ inline void bitonic_sort_desc(unsigned long long* keys, int n,
                                         int tid, int nt)
{
  for (int size = 2; size <= n; size <<= 1) {
    for (int stride = size >> 1; stride > 0; stride >>= 1) {
      __syncthreads();
      for (int i = tid; i < n; i += nt) {
        const int j = i ^ stride;
        if (j > i) {
          unsigned long long x = keys[i], y = keys[j];
          if ((x < y) == ((i & size) == 0)) { keys[i] = y; keys[j] = x; }
        }
      }
    }
  }
  __syncthreads();
}

// ---------------------------------------------------------------------------
// Kernel 2: exact deterministic top-1000 prefilter for a level (one block per
// batch). Full bitonic sort of padded keys in LDS (320KB/WGP easily fits the
// 64KB level-0 sort), then gather boxes+scores into the candidate arrays.
// ---------------------------------------------------------------------------
__global__ void select_topk_kernel(
    const float* __restrict__ dboxes, const float* __restrict__ dscores,
    const float* __restrict__ dmax,
    float* __restrict__ cboxes, float* __restrict__ cscores,
    int n, int padN, int lvl_off, int cand_off, int take)
{
  extern __shared__ unsigned long long keys[];
  const int b = blockIdx.x, tid = threadIdx.x, nt = blockDim.x;

  for (int i = tid; i < padN; i += nt) {
    unsigned long long k = 0ull;
    if (i < n) {
      const float s = dmax[(size_t)b * NA_TOT + lvl_off + i];
      const unsigned u = __float_as_uint(s) ^ 0x80000000u;  // s > 0 always
      k = ((unsigned long long)u << 32) | (unsigned)(0xFFFFFFFFu - (unsigned)i);
    }
    keys[i] = k;
  }
  bitonic_sort_desc(keys, padN, tid, nt);

  for (int slot = tid; slot < take; slot += nt) {
    const int a = (int)(0xFFFFFFFFu - (unsigned)keys[slot]);
    const float* sb = dboxes  + ((size_t)b * NA_TOT + lvl_off + a) * 4;
    float*       db = cboxes  + ((size_t)b * NCAND + cand_off + slot) * 4;
    *(float4*)db = *(const float4*)sb;
    const float* ss = dscores + ((size_t)b * NA_TOT + lvl_off + a) * NCLS;
    float*       ds = cscores + ((size_t)b * NCAND + cand_off + slot) * NCLS;
#pragma unroll
    for (int c4 = 0; c4 < 5; ++c4) ((float4*)ds)[c4] = ((const float4*)ss)[c4];
  }
}

// Level 2 (400 anchors <= NMS_PRE): straight copy, original order preserved.
__global__ void copy_level2_kernel(
    const float* __restrict__ dboxes, const float* __restrict__ dscores,
    float* __restrict__ cboxes, float* __restrict__ cscores)
{
  const int b = blockIdx.x, tid = threadIdx.x;
  for (int slot = tid; slot < 400; slot += blockDim.x) {
    const float* sb = dboxes  + ((size_t)b * NA_TOT + 8000 + slot) * 4;
    float*       db = cboxes  + ((size_t)b * NCAND + 2000 + slot) * 4;
    *(float4*)db = *(const float4*)sb;
    const float* ss = dscores + ((size_t)b * NA_TOT + 8000 + slot) * NCLS;
    float*       ds = cscores + ((size_t)b * NCAND + 2000 + slot) * NCLS;
#pragma unroll
    for (int c4 = 0; c4 < 5; ++c4) ((float4*)ds)[c4] = ((const float4*)ss)[c4];
  }
}

// ---------------------------------------------------------------------------
// Kernel 3: greedy per-class NMS. One block per (batch, class).
// Boxes + scores staged into LDS with CDNA5 async global->LDS transfers
// (~48KB resident). 100 fixed iterations: block-wide argmax (lowest-index
// tie-break == jnp.argmax), IoU suppression.
// ---------------------------------------------------------------------------
__global__ __launch_bounds__(256) void nms_kernel(
    const float* __restrict__ cboxes, const float* __restrict__ cscores,
    float* __restrict__ sel_s, float* __restrict__ sel_b)
{
  __shared__ __align__(16) float sb[NCAND * 4];
  __shared__ float ss[NCAND];
  __shared__ float rs[256];
  __shared__ int   ri[256];

  const int bc = blockIdx.x;
  const int b  = bc / NCLS;
  const int c  = bc % NCLS;
  const int tid = threadIdx.x;

  // Async stage: boxes as per-lane B128 (16B) transfers, per-class scores as
  // B32 gather (stride 80B). ASYNCcnt fenced, then workgroup barrier.
  for (int i = tid; i < NCAND; i += 256) {
    async_g2lds_b128(&sb[i * 4], cboxes + ((size_t)b * NCAND + i) * 4);
    async_g2lds_b32(&ss[i], cscores + ((size_t)b * NCAND + i) * NCLS + c);
  }
  wait_async0();
  __syncthreads();

  for (int t = 0; t < TOPK; ++t) {
    // block argmax with first-occurrence tie-break
    float best = -__builtin_inff();
    int   bi   = 0x7fffffff;
    for (int i = tid; i < NCAND; i += 256) {
      const float s = ss[i];
      if (s > best || (s == best && i < bi)) { best = s; bi = i; }
    }
    rs[tid] = best; ri[tid] = bi;
    for (int off = 128; off > 0; off >>= 1) {
      __syncthreads();
      if (tid < off) {
        const float s2 = rs[tid + off]; const int i2 = ri[tid + off];
        if (s2 > rs[tid] || (s2 == rs[tid] && i2 < ri[tid])) {
          rs[tid] = s2; ri[tid] = i2;
        }
      }
    }
    __syncthreads();
    const float bs   = rs[0];
    const int   bidx = ri[0];
    const float by1 = sb[bidx * 4 + 0], bx1 = sb[bidx * 4 + 1];
    const float by2 = sb[bidx * 4 + 2], bx2 = sb[bidx * 4 + 3];
    if (tid == 0) {
      sel_s[(size_t)bc * TOPK + t] = bs;
      float4 ob; ob.x = by1; ob.y = bx1; ob.z = by2; ob.w = bx2;
      *(float4*)(sel_b + ((size_t)bc * TOPK + t) * 4) = ob;
    }
    const float a1 = (by2 - by1) * (bx2 - bx1);
    __syncthreads();
    // suppression
    for (int i = tid; i < NCAND; i += 256) {
      if (i == bidx) { ss[i] = -__builtin_inff(); continue; }
      const float y1 = sb[i * 4 + 0], x1 = sb[i * 4 + 1];
      const float y2 = sb[i * 4 + 2], x2 = sb[i * 4 + 3];
      const float yy1 = fmaxf(by1, y1), xx1 = fmaxf(bx1, x1);
      const float yy2 = fminf(by2, y2), xx2 = fminf(bx2, x2);
      const float inter = fmaxf(yy2 - yy1, 0.0f) * fmaxf(xx2 - xx1, 0.0f);
      const float a2 = (y2 - y1) * (x2 - x1);
      const float iou = inter / (a1 + a2 - inter + 1e-9f);
      if (iou > IOU_T) ss[i] = -__builtin_inff();
    }
    __syncthreads();
  }
}

// ---------------------------------------------------------------------------
// Kernel 4: global top-100 over the 20x100 per-class picks + score mask.
// One block per batch; 2048-wide bitonic sort in LDS (flat-index tie-break
// matches tf/jax top_k over the reshaped [C*K] array).
// ---------------------------------------------------------------------------
__global__ __launch_bounds__(256) void combine_kernel(
    const float* __restrict__ sel_s, const float* __restrict__ sel_b,
    float* __restrict__ out)
{
  __shared__ unsigned long long keys[2048];
  const int b = blockIdx.x, tid = threadIdx.x;

  for (int i = tid; i < 2048; i += 256) {
    unsigned long long k = 0ull;
    if (i < NCLS * TOPK) {
      float s = sel_s[(size_t)b * NCLS * TOPK + i];
      if (!(s > -1e37f)) s = -1.0f;              // -inf -> -1.0 (reference)
      const unsigned bits = __float_as_uint(s);
      const unsigned u = (bits & 0x80000000u) ? ~bits : (bits | 0x80000000u);
      k = ((unsigned long long)u << 32) | (unsigned)(0xFFFFFFFFu - (unsigned)i);
    }
    keys[i] = k;
  }
  bitonic_sort_desc(keys, 2048, tid, 256);

  if (tid < TOPK) {
    const unsigned long long k = keys[tid];
    const unsigned u = (unsigned)(k >> 32);
    const unsigned bits = (u & 0x80000000u) ? (u & 0x7FFFFFFFu) : ~u;
    const float s = __uint_as_float(bits);
    const unsigned flat = 0xFFFFFFFFu - (unsigned)k;
    float* o = out + ((size_t)b * TOPK + tid) * 6;
    if (s > SCORE_T) {                           // score mask (> 0.3)
      const float4 bx = *(const float4*)(sel_b + ((size_t)b * NCLS * TOPK + flat) * 4);
      o[0] = bx.x; o[1] = bx.y; o[2] = bx.z; o[3] = bx.w;
      o[4] = s;    o[5] = (float)(flat / TOPK);
    } else {
      o[0] = 0.f; o[1] = 0.f; o[2] = 0.f; o[3] = 0.f; o[4] = 0.f; o[5] = 0.f;
    }
  }
}

// ---------------------------------------------------------------------------
extern "C" void kernel_launch(void* const* d_in, const int* in_sizes, int n_in,
                              void* d_out, int out_size, void* d_ws, size_t ws_size,
                              hipStream_t stream)
{
  (void)in_sizes; (void)n_in; (void)out_size; (void)ws_size;
  // setup_inputs() order: cls0, bbox0, cls1, bbox1, cls2, bbox2, origin_shapes
  const float* cls0  = (const float*)d_in[0];
  const float* bbox0 = (const float*)d_in[1];
  const float* cls1  = (const float*)d_in[2];
  const float* bbox1 = (const float*)d_in[3];
  const float* cls2  = (const float*)d_in[4];
  const float* bbox2 = (const float*)d_in[5];
  // d_in[6] (origin_shapes) is unused by the reference computation.

  char* ws = (char*)d_ws;                       // all offsets 16B-aligned
  float* dboxes  = (float*)(ws + 0);            // 16*8400*4   f32
  float* dscores = (float*)(ws + 2150400);      // 16*8400*20  f32
  float* dmax    = (float*)(ws + 12902400);     // 16*8400     f32
  float* cboxes  = (float*)(ws + 13440000);     // 16*2400*4   f32
  float* cscores = (float*)(ws + 14054400);     // 16*2400*20  f32
  float* sel_s   = (float*)(ws + 17126400);     // 16*20*100   f32
  float* sel_b   = (float*)(ws + 17254400);     // 16*20*100*4 f32

  // decode: 16 anchors/wave, 8 waves/block
  decode_level_kernel<<<800, 256, 0, stream>>>(cls0, bbox0, dboxes, dscores, dmax,
                                               80, 80, 8.0f,  0,    400);
  decode_level_kernel<<<200, 256, 0, stream>>>(cls1, bbox1, dboxes, dscores, dmax,
                                               40, 40, 16.0f, 6400, 100);
  decode_level_kernel<<< 50, 256, 0, stream>>>(cls2, bbox2, dboxes, dscores, dmax,
                                               20, 20, 32.0f, 8000, 25);

  // exact top-1000 prefilter (levels 0 and 1), level 2 copied verbatim
  select_topk_kernel<<<16, 1024, 8192 * 8, stream>>>(dboxes, dscores, dmax,
                                                     cboxes, cscores,
                                                     6400, 8192, 0,    0,    1000);
  select_topk_kernel<<<16, 1024, 2048 * 8, stream>>>(dboxes, dscores, dmax,
                                                     cboxes, cscores,
                                                     1600, 2048, 6400, 1000, 1000);
  copy_level2_kernel<<<16, 256, 0, stream>>>(dboxes, dscores, cboxes, cscores);

  // per-(batch,class) NMS, then global top-100 + mask
  nms_kernel<<<BATCH * NCLS, 256, 0, stream>>>(cboxes, cscores, sel_s, sel_b);
  combine_kernel<<<BATCH, 256, 0, stream>>>(sel_s, sel_b, (float*)d_out);
}
